// LSTM_38920993636832
// MI455X (gfx1250) — matrix operand order repaired and checked by reading, not compiled, not run on hardware
//
#include <hip/hip_runtime.h>

// LSTM: B=1024, T=512, I=64, H=256, O=1
#define BATCH 1024
#define TSEQ  512
#define ISZ   64
#define HSZ   256
#define KTOT  320                 // ISZ + HSZ
#define NKT   10                  // KTOT / 32 (K-tiles per step)
#define NNT   64                  // 4H / 16  (N-tiles total)
#define AROW  336                 // padded LDS row in ushorts (672B, 16B multiple)
#define FRAG_USHORTS 512          // 32 lanes * 16 bf16 per B fragment

typedef __attribute__((ext_vector_type(16))) __bf16        v16bf;
typedef __attribute__((ext_vector_type(8)))  float         v8f;
typedef __attribute__((ext_vector_type(4)))  unsigned int  u32x4;
typedef __attribute__((ext_vector_type(4)))  float         f32x4;
typedef __attribute__((ext_vector_type(4)))  unsigned short u16x4;

union FragU { u32x4 q[2]; v16bf v; };

__device__ __forceinline__ unsigned short f2bf(float f) {
    unsigned int u = __float_as_uint(f);
    u += 0x7FFFu + ((u >> 16) & 1u);          // round-to-nearest-even
    return (unsigned short)(u >> 16);
}
__device__ __forceinline__ float bf2f(unsigned short b) {
    return __uint_as_float(((unsigned int)b) << 16);
}
__device__ __forceinline__ float sigmoidf_fast(float x) {
    return 1.0f / (1.0f + __expf(-x));
}

// ---------------------------------------------------------------------------
// Prep kernel 1: pack W_ih (4H x I) and W_hh (4H x H) as bf16 in WMMA
// B-operand fragment order: wpack[nt*NKT + kt][lane][16]
//   B[k, n] with n = nt*16 + (lane&15), k = kt*32 + (lane>=16 ? 16 : 0) + j
// ---------------------------------------------------------------------------
__global__ void lstm_pack_weights(const float* __restrict__ W_ih,
                                  const float* __restrict__ W_hh,
                                  unsigned short* __restrict__ wpack) {
    const int frag  = blockIdx.x;             // 0..639  (= nt*NKT + kt)
    const int nt    = frag / NKT;
    const int kt    = frag % NKT;
    const int lane  = threadIdx.x;            // 0..31
    const int n     = nt * 16 + (lane & 15);
    const int kbase = kt * 32 + ((lane >> 4) << 4);
    unsigned short* dst = wpack + (size_t)frag * FRAG_USHORTS + lane * 16;
    #pragma unroll
    for (int j = 0; j < 16; ++j) {
        const int k = kbase + j;
        const float w = (k < ISZ) ? W_ih[n * ISZ + k]
                                  : W_hh[n * HSZ + (k - ISZ)];
        dst[j] = f2bf(w);
    }
}

// Prep kernel 2: bsum[n] = b_ih[n] + b_hh[n]
__global__ void lstm_pack_bias(const float* __restrict__ b_ih,
                               const float* __restrict__ b_hh,
                               float* __restrict__ bsum) {
    const int i = blockIdx.x * blockDim.x + threadIdx.x;
    if (i < 4 * HSZ) bsum[i] = b_ih[i] + b_hh[i];
}

// ---------------------------------------------------------------------------
// Main recurrent kernel. One workgroup = 16 batch rows, 8 waves.
// Wave w owns hidden columns [w*32, w*32+32): computes i,f,g,o for that slice.
// B fragments stream from L2 every step. The per-iteration offset is
// laundered (not the pointer!), so loads stay global_load_b128 (GVS mode,
// LOADcnt only) and cannot be hoisted across the recurrence and spilled.
// ---------------------------------------------------------------------------
__global__ __launch_bounds__(256, 1) void lstm_recurrent(
    const float* __restrict__ x,              // [B, T, I] f32
    const unsigned short* __restrict__ wpack, // packed bf16 B fragments
    const float* __restrict__ bsum,           // [4H]
    const float* __restrict__ W_out,          // [1, H]
    const float* __restrict__ b_out,          // [1]
    float* __restrict__ out)                  // [B]
{
    __shared__ unsigned short Abuf[16][AROW]; // cols 0..63 = x_t, 64..319 = h
    __shared__ float outacc[16];

    const int tid    = threadIdx.x;
    const int wave   = tid >> 5;
    const int lane   = tid & 31;
    const int laneN  = lane & 15;
    const int laneHi = lane >> 4;             // 0 | 1
    const int mbase  = laneHi << 3;           // C/D row base (0 | 8)
    const int b0     = blockIdx.x * 16;

    // per-(wave,lane) base element offset into wpack; per-(g,t2,kt) deltas are
    // compile-time constants after unrolling -> single-immediate global loads
    const int boff = wave * (2 * NKT * FRAG_USHORTS) + lane * 16;

    // zero the h region (cols 64..335 incl. padding); x region staged below
    for (int i = tid; i < 16 * (AROW - ISZ); i += 256) {
        const int m = i / (AROW - ISZ);
        const int c = ISZ + (i - m * (AROW - ISZ));
        Abuf[m][c] = 0;
    }

    // per-lane bias values for accumulator seeding
    float biasv[4][2];
    #pragma unroll
    for (int g = 0; g < 4; ++g)
        #pragma unroll
        for (int t2 = 0; t2 < 2; ++t2)
            biasv[g][t2] = bsum[g * HSZ + wave * 32 + t2 * 16 + laneN];

    v8f acc[4][2];
    v8f cst[2];
    #pragma unroll
    for (int t2 = 0; t2 < 2; ++t2) {
        #pragma unroll
        for (int r = 0; r < 8; ++r) cst[t2][r] = 0.0f;
        #pragma unroll
        for (int g = 0; g < 4; ++g)
            #pragma unroll
            for (int r = 0; r < 8; ++r) acc[g][t2][r] = biasv[g][t2];
    }

    // stage x(t=0): 16 rows x 64 cols, 4 contiguous f32 per thread
    {
        const int e = tid * 4;
        const int m = e >> 6, k = e & 63;
        const f32x4 xv = __builtin_nontemporal_load(
            (const f32x4*)&x[((size_t)(b0 + m) * TSEQ + 0) * ISZ + k]);
        u16x4 s;
        s[0] = f2bf(xv[0]); s[1] = f2bf(xv[1]); s[2] = f2bf(xv[2]); s[3] = f2bf(xv[3]);
        *(u16x4*)&Abuf[m][k] = s;
    }
    __syncthreads();

    for (int t = 0; t < TSEQ; ++t) {
        // Launder the 32-bit element offset: value unchanged, but unknowable
        // per iteration -> B loads cannot be hoisted out of the t loop, while
        // the base pointer keeps global address-space provenance
        // (global_load_b128, saddr+voffset, LOADcnt only).
        int bofft = boff;
        asm volatile("" : "+v"(bofft));

        // ---- gates = [x_t | h] @ W^T + b : 10 K-tiles x 8 N-tiles of WMMA ----
        #pragma unroll
        for (int kt = 0; kt < NKT; ++kt) {
            FragU a;
            const int kb = kt * 32 + (laneHi << 4);
            a.q[0] = *(const u32x4*)&Abuf[laneN][kb];
            a.q[1] = *(const u32x4*)&Abuf[laneN][kb + 8];
            #pragma unroll
            for (int g = 0; g < 4; ++g) {
                #pragma unroll
                for (int t2 = 0; t2 < 2; ++t2) {
                    // frag index = (g*16 + wave*2 + t2)*NKT + kt
                    const int delta = g * (16 * NKT * FRAG_USHORTS)
                                    + t2 * (NKT * FRAG_USHORTS)
                                    + kt * FRAG_USHORTS;
                    FragU b;
                    const u32x4* bp = (const u32x4*)(wpack + bofft + delta);
                    b.q[0] = bp[0];
                    b.q[1] = bp[1];
                    acc[g][t2] = __builtin_amdgcn_wmma_f32_16x16x32_bf16(
                        false, a.v, false, b.v, (short)0, acc[g][t2],
                        false, false);
                }
            }
        }
        __syncthreads();   // all waves done reading Abuf for this step

        // ---- activations, cell update, write new h, reseed accumulators ----
        #pragma unroll
        for (int t2 = 0; t2 < 2; ++t2) {
            const int ncol = wave * 32 + t2 * 16 + laneN;
            #pragma unroll
            for (int r = 0; r < 8; ++r) {
                const float ig = sigmoidf_fast(acc[0][t2][r]);
                const float fg = sigmoidf_fast(acc[1][t2][r]);
                const float gg = tanhf(acc[2][t2][r]);
                const float og = sigmoidf_fast(acc[3][t2][r]);
                const float c  = fg * cst[t2][r] + ig * gg;
                cst[t2][r] = c;
                const float h = og * tanhf(c);
                Abuf[mbase + r][ISZ + ncol] = f2bf(h);
            }
            #pragma unroll
            for (int g = 0; g < 4; ++g)
                #pragma unroll
                for (int r = 0; r < 8; ++r) acc[g][t2][r] = biasv[g][t2];
        }

        // stage x(t+1) while h-writes land (disjoint LDS regions)
        if (t + 1 < TSEQ) {
            const int e = tid * 4;
            const int m = e >> 6, k = e & 63;
            const f32x4 xv = __builtin_nontemporal_load(
                (const f32x4*)&x[((size_t)(b0 + m) * TSEQ + (t + 1)) * ISZ + k]);
            u16x4 s;
            s[0] = f2bf(xv[0]); s[1] = f2bf(xv[1]);
            s[2] = f2bf(xv[2]); s[3] = f2bf(xv[3]);
            *(u16x4*)&Abuf[m][k] = s;
        }
        __syncthreads();
    }

    // ---- output projection: out[b] = h_last[b,:] . W_out + b_out ----
    if (tid < 16) outacc[tid] = b_out[0];
    __syncthreads();
    #pragma unroll
    for (int t2 = 0; t2 < 2; ++t2) {
        const int ncol = wave * 32 + t2 * 16 + laneN;
        const float wv = W_out[ncol];
        #pragma unroll
        for (int r = 0; r < 8; ++r) {
            const float h = bf2f(Abuf[mbase + r][ISZ + ncol]);
            atomicAdd(&outacc[mbase + r], h * wv);
        }
    }
    __syncthreads();
    if (tid < 16) out[b0 + tid] = outacc[tid];
}

// ---------------------------------------------------------------------------
extern "C" void kernel_launch(void* const* d_in, const int* in_sizes, int n_in,
                              void* d_out, int out_size, void* d_ws, size_t ws_size,
                              hipStream_t stream) {
    const float* x     = (const float*)d_in[0];   // [1024, 512, 64]
    const float* W_ih  = (const float*)d_in[1];   // [1024, 64]
    const float* W_hh  = (const float*)d_in[2];   // [1024, 256]
    const float* b_ih  = (const float*)d_in[3];   // [1024]
    const float* b_hh  = (const float*)d_in[4];   // [1024]
    const float* W_out = (const float*)d_in[5];   // [1, 256]
    const float* b_out = (const float*)d_in[6];   // [1]
    float* out = (float*)d_out;                   // [1024]

    // workspace layout: packed bf16 weights (655360 B) | bias sum (4096 B)
    unsigned short* wpack = (unsigned short*)d_ws;
    float* bsum = (float*)((char*)d_ws + (size_t)NNT * NKT * FRAG_USHORTS * 2);

    lstm_pack_weights<<<NNT * NKT, 32, 0, stream>>>(W_ih, W_hh, wpack);
    lstm_pack_bias<<<4, 256, 0, stream>>>(b_ih, b_hh, bsum);
    lstm_recurrent<<<BATCH / 16, 256, 0, stream>>>(x, wpack, bsum, W_out, b_out, out);
}
